// _PixelCNNDistribution_27178553049953
// MI455X (gfx1250) — compile-verified, matrix operand order checked by reading, not run on hardware
//
#include <hip/hip_runtime.h>
#include <hip/hip_bf16.h>

typedef __attribute__((ext_vector_type(16))) _Float16 v16h;
typedef __attribute__((ext_vector_type(8)))  _Float16 v8h;
typedef __attribute__((ext_vector_type(8)))  float    v8f;

#define NPOS  32768   // B*H*W
#define FCH   160
#define C2F   320
#define KC10  10      // 320/32

__device__ __forceinline__ float eluf(float v)      { return v > 0.f ? v : expf(v) - 1.f; }
__device__ __forceinline__ float sigmf(float v)     { return 1.f / (1.f + expf(-v)); }
__device__ __forceinline__ float softplusf(float v) { return fmaxf(v, 0.f) + log1pf(expf(-fabsf(v))); }

// ---------------------------------------------------------------------------
// Build xp = concat(2*samples-1, ones)  ->  [pos][4] fp32
// ---------------------------------------------------------------------------
__global__ void k_prep(const float* __restrict__ s, float* __restrict__ xp) {
    int tid = blockIdx.x * blockDim.x + threadIdx.x;
    if (tid >= NPOS * 4) return;
    int c = tid & 3, pos = tid >> 2;
    int b = pos >> 10, hw = pos & 1023;
    float v = (c == 3) ? 1.f : 2.f * s[((size_t)(b * 3 + c) << 10) + hw] - 1.f;
    xp[tid] = v;
}

// ---------------------------------------------------------------------------
// Tiny init convs (Cin=4) with optional down/right shift and accumulation.
// ---------------------------------------------------------------------------
__global__ void k_init_conv(const float* __restrict__ xp, const float* __restrict__ w,
                            const float* __restrict__ bias, float* __restrict__ out,
                            int KH, int KW, int padT, int padL,
                            int shiftD, int shiftR, int accum) {
    int tid = blockIdx.x * blockDim.x + threadIdx.x;
    if (tid >= NPOS * FCH) return;
    int o = tid % FCH, pos = tid / FCH;
    int b = pos >> 10, h = (pos >> 5) & 31, wq = pos & 31;
    int he = h - shiftD, we = wq - shiftR;
    float val = 0.f;
    if (he >= 0 && we >= 0) {
        val = bias[o];
        for (int i = 0; i < KH; i++)
            for (int j = 0; j < KW; j++) {
                int hs = he - padT + i, ws = we - padL + j;
                if (hs < 0 || hs >= 32 || ws < 0 || ws >= 32) continue;
                const float* xv = xp + (((size_t)(b * 32 + hs) * 32 + ws) * 4);
                for (int c = 0; c < 4; c++)
                    val += xv[c] * w[((o * 4 + c) * KH + i) * KW + j];
            }
    }
    size_t idx = (size_t)pos * FCH + o;
    out[idx] = accum ? out[idx] + val : val;
}

// ---------------------------------------------------------------------------
// concat_elu: fp32 [pos][160] -> f16 [pos][320]
// ---------------------------------------------------------------------------
__global__ void k_celu(const float* __restrict__ in, _Float16* __restrict__ out) {
    int tid = blockIdx.x * blockDim.x + threadIdx.x;
    if (tid >= NPOS * FCH) return;
    int pos = tid / FCH, c = tid % FCH;
    float v = in[tid];
    out[(size_t)pos * C2F + c]       = (_Float16)eluf(v);
    out[(size_t)pos * C2F + FCH + c] = (_Float16)eluf(-v);
}

// ---------------------------------------------------------------------------
// Gate: x += g1 * sigmoid(g2); also emit concat_elu(x_new) f16 for next conv
// ---------------------------------------------------------------------------
__global__ void k_gate(float* __restrict__ x, const float* __restrict__ c2,
                       _Float16* __restrict__ ceout) {
    int tid = blockIdx.x * blockDim.x + threadIdx.x;
    if (tid >= NPOS * FCH) return;
    int pos = tid / FCH, c = tid % FCH;
    float g1 = c2[(size_t)pos * C2F + c];
    float g2 = c2[(size_t)pos * C2F + FCH + c];
    float v = x[tid] + g1 * sigmf(g2);
    x[tid] = v;
    ceout[(size_t)pos * C2F + c]       = (_Float16)eluf(v);
    ceout[(size_t)pos * C2F + FCH + c] = (_Float16)eluf(-v);
}

// ---------------------------------------------------------------------------
// Pack OIHW fp32 weights into per-lane WMMA B-fragment layout (f16):
// flat = (((ct*taps + t)*KC + kc)*32 + lane)*16 + e
// ---------------------------------------------------------------------------
__global__ void k_pack(const float* __restrict__ w, _Float16* __restrict__ wpk,
                       int Cout, int KH, int KW) {
    int taps = KH * KW;
    int total = Cout * C2F * taps;
    int tid = blockIdx.x * blockDim.x + threadIdx.x;
    if (tid >= total) return;
    int e  = tid & 15;
    int l  = (tid >> 4) & 31;
    int kc = (tid >> 9) % KC10;
    int t  = (tid / (512 * KC10)) % taps;
    int ct = tid / (512 * KC10 * taps);
    int n = l & 15, khalf = l >> 4;
    int kk = (e < 8) ? (khalf * 8 + e) : (16 + khalf * 8 + (e - 8));
    int cin = kc * 32 + kk;
    int cout = ct * 16 + n;
    int i = t / KW, j = t % KW;
    float v = w[(((size_t)cout * C2F + cin) * KH + i) * KW + j];
    wpk[tid] = (_Float16)v;
}

// ---------------------------------------------------------------------------
// WMMA implicit-GEMM conv. Cin=320 f16 NHWC input; Cout in {160,320}.
// Wave tile: 32 positions (one image row) x 80 out-channels.
// All 8 waves of a workgroup share the same 80-channel group, so the current
// tap's weight slice (5 tiles x 10 kc x 1KB = 50KB) is staged once into LDS.
// ---------------------------------------------------------------------------
__global__ void __launch_bounds__(256)
k_conv_wmma(const _Float16* __restrict__ ce, const _Float16* __restrict__ wpk,
            const float* __restrict__ bias, float* __restrict__ out,
            int Cout, int KH, int KW, int padT, int padL, int beta) {
    __shared__ __align__(16) _Float16 sw[25600];   // 50 KB weight stage
    const int tid  = threadIdx.x;
    const int lane = tid & 31;
    const int wid  = (blockIdx.x * blockDim.x + tid) >> 5;
    const int row  = wid & 1023;          // b*32 + h   (8 consecutive rows / block)
    const int cg   = wid >> 10;           // 80-channel group (uniform per block)
    const int b = row >> 5, h = row & 31;
    const int m = lane & 15, khalf = lane >> 4;
    const int taps = KH * KW;

    v8f acc[2][5];
    const v8f vzero = {0.f, 0.f, 0.f, 0.f, 0.f, 0.f, 0.f, 0.f};
    for (int pt = 0; pt < 2; pt++)
        for (int ci = 0; ci < 5; ci++) acc[pt][ci] = vzero;

    for (int t = 0; t < taps; t++) {
        // ---- stage this tap's 80-cout weight slice into LDS ----
        __syncthreads();                   // previous tap's reads complete
        for (int q = tid; q < 3200; q += 256) {      // 3200 x 16B chunks
            int ci = q / 640, r = q - ci * 640;      // per-ci slice is 5120 halfs
            const v8h* src = (const v8h*)(wpk + (((size_t)(cg * 5 + ci) * taps + t) * 5120));
            *((v8h*)sw + (size_t)ci * 640 + r) = src[r];
        }
        __syncthreads();
        // prefetch a slice of the next tap's weights into GL2 while we compute
        if (t + 1 < taps && tid < 80) {
            const char* nxt = (const char*)(wpk + (((size_t)(cg * 5) * taps + (t + 1)) * 5120));
            __builtin_prefetch(nxt + tid * 128, 0, 1);
        }

        int i = t / KW, j = t % KW;
        int hs = h + i - padT;
        bool rok = (hs >= 0) && (hs < 32);
        int dw = j - padL;
        const _Float16* ap[2];
        bool aok[2];
        for (int pt = 0; pt < 2; pt++) {
            int ws = pt * 16 + m + dw;
            aok[pt] = rok && ws >= 0 && ws < 32;
            ap[pt] = aok[pt] ? (ce + ((size_t)((b * 32 + hs) * 32 + ws)) * C2F) : ce;
        }
        for (int kc = 0; kc < KC10; kc++) {
            v16h afrag[2];
#pragma unroll
            for (int pt = 0; pt < 2; pt++) {
                v8h lo = {0, 0, 0, 0, 0, 0, 0, 0};
                v8h hi = {0, 0, 0, 0, 0, 0, 0, 0};
                if (aok[pt]) {
                    lo = *(const v8h*)(ap[pt] + kc * 32 + khalf * 8);
                    hi = *(const v8h*)(ap[pt] + kc * 32 + 16 + khalf * 8);
                }
                afrag[pt] = __builtin_shufflevector(lo, hi, 0, 1, 2, 3, 4, 5, 6, 7,
                                                    8, 9, 10, 11, 12, 13, 14, 15);
            }
#pragma unroll
            for (int ci = 0; ci < 5; ci++) {
                const v16h bfrag = *(const v16h*)(sw + ((size_t)(ci * KC10 + kc) * 512) + lane * 16);
                acc[0][ci] = __builtin_amdgcn_wmma_f32_16x16x32_f16(
                    false, afrag[0], false, bfrag, (short)0, acc[0][ci], false, false);
                acc[1][ci] = __builtin_amdgcn_wmma_f32_16x16x32_f16(
                    false, afrag[1], false, bfrag, (short)0, acc[1][ci], false, false);
            }
        }
    }

    // Epilogue: D layout -> lanes 0-15: M=r,N=lane ; lanes 16-31: M=r+8,N=lane-16
    const int n = lane & 15;
    const int mofs = (lane >> 4) * 8;
    for (int ci = 0; ci < 5; ci++) {
        int c = (cg * 5 + ci) * 16 + n;
        float bv = bias ? bias[c] : 0.f;
        for (int pt = 0; pt < 2; pt++) {
#pragma unroll
            for (int r = 0; r < 8; r++) {
                int wq = pt * 16 + r + mofs;
                size_t idx = (((size_t)(b * 32 + h) * 32) + wq) * Cout + c;
                float v = acc[pt][ci][r] + bv;
                out[idx] = beta ? (out[idx] + v) : v;
            }
        }
    }
}

// ---------------------------------------------------------------------------
// elu(ul) fp32, then NIN to P=100 params
// ---------------------------------------------------------------------------
__global__ void k_elu(const float* __restrict__ in, float* __restrict__ out) {
    int tid = blockIdx.x * blockDim.x + threadIdx.x;
    if (tid >= NPOS * FCH) return;
    out[tid] = eluf(in[tid]);
}

__global__ void k_ninout(const float* __restrict__ eul, const float* __restrict__ w,
                         const float* __restrict__ bias, float* __restrict__ params) {
    int tid = blockIdx.x * blockDim.x + threadIdx.x;
    if (tid >= NPOS * 100) return;
    int pos = tid / 100, p = tid % 100;
    const float* xr = eul + (size_t)pos * FCH;
    const float* wr = w + (size_t)p * FCH;
    float s = bias[p];
    for (int c = 0; c < FCH; c++) s += xr[c] * wr[c];
    params[tid] = s;
}

__global__ void k_zero(float* out) {
    if (threadIdx.x == 0 && blockIdx.x == 0) out[0] = 0.f;
}

// ---------------------------------------------------------------------------
// Discretized mixture-of-logistics; final = +sum_pos logsumexp_m(...)
// ---------------------------------------------------------------------------
__device__ __forceinline__ float lpchan(float xv, float mean, float ls) {
    float centered = xv - mean;
    float inv = expf(-ls);
    float plus_in = inv * (centered + 1.f / 255.f);
    float min_in  = inv * (centered - 1.f / 255.f);
    float cdf_delta = sigmf(plus_in) - sigmf(min_in);
    float log_cdf_plus = plus_in - softplusf(plus_in);
    float log_1m = -softplusf(min_in);
    float mid = inv * centered;
    float log_pdf_mid = mid - ls - 2.f * softplusf(mid);
    float r;
    if (xv < -0.999f)            r = log_cdf_plus;
    else if (xv > 0.999f)        r = log_1m;
    else if (cdf_delta > 1e-5f)  r = logf(fmaxf(cdf_delta, 1e-12f));
    else                         r = log_pdf_mid - 4.8481164f;  // log(127.5)
    return r;
}

__global__ void __launch_bounds__(256)
k_dmll(const float* __restrict__ samples, const float* __restrict__ params,
       float* __restrict__ out) {
    int pos = blockIdx.x * blockDim.x + threadIdx.x;   // exactly NPOS threads
    int b = pos >> 10, hw = pos & 1023;
    float xs[3];
    for (int c = 0; c < 3; c++)
        xs[c] = 2.f * samples[((size_t)(b * 3 + c) << 10) + hw] - 1.f;
    const float* l = params + (size_t)pos * 100;

    float lg[10], mx = -1e30f;
    for (int k = 0; k < 10; k++) { lg[k] = l[k]; mx = fmaxf(mx, lg[k]); }
    float se = 0.f;
    for (int k = 0; k < 10; k++) se += expf(lg[k] - mx);
    float lse_logit = mx + logf(se);

    float terms[10], mxt = -1e30f;
    for (int k = 0; k < 10; k++) {
        float mean0 = l[10 + k], ls0 = fmaxf(l[20 + k], -7.f), kA = tanhf(l[30 + k]);
        float mean1 = l[40 + k], ls1 = fmaxf(l[50 + k], -7.f), kB = tanhf(l[60 + k]);
        float mean2 = l[70 + k], ls2 = fmaxf(l[80 + k], -7.f), kC = tanhf(l[90 + k]);
        float m2 = mean1 + kA * xs[0];
        float m3 = mean2 + kB * xs[0] + kC * xs[1];
        float t = lpchan(xs[0], mean0, ls0) + lpchan(xs[1], m2, ls1) +
                  lpchan(xs[2], m3, ls2) + (lg[k] - lse_logit);
        terms[k] = t;
        mxt = fmaxf(mxt, t);
    }
    float s2 = 0.f;
    for (int k = 0; k < 10; k++) s2 += expf(terms[k] - mxt);
    float val = mxt + logf(s2);

    __shared__ float red[256];
    red[threadIdx.x] = val;
    __syncthreads();
    for (int s = 128; s > 0; s >>= 1) {
        if (threadIdx.x < s) red[threadIdx.x] += red[threadIdx.x + s];
        __syncthreads();
    }
    if (threadIdx.x == 0) atomicAdd(out, red[0]);
}

// ---------------------------------------------------------------------------
extern "C" void kernel_launch(void* const* d_in, const int* in_sizes, int n_in,
                              void* d_out, int out_size, void* d_ws, size_t ws_size,
                              hipStream_t stream) {
    const float* samples  = (const float*)d_in[0];
    const float* w_u_init = (const float*)d_in[1];
    const float* b_u_init = (const float*)d_in[2];
    const float* w_ul_d   = (const float*)d_in[3];
    const float* b_ul_d   = (const float*)d_in[4];
    const float* w_ul_dr  = (const float*)d_in[5];
    const float* b_ul_dr  = (const float*)d_in[6];
    const float* u_c1_w   = (const float*)d_in[7];
    const float* u_c1_b   = (const float*)d_in[8];
    const float* u_c2_w   = (const float*)d_in[9];
    const float* u_c2_b   = (const float*)d_in[10];
    const float* ul_c1_w  = (const float*)d_in[11];
    const float* ul_c1_b  = (const float*)d_in[12];
    const float* ul_nin_w = (const float*)d_in[13];
    const float* ul_nin_b = (const float*)d_in[14];
    const float* ul_c2_w  = (const float*)d_in[15];
    const float* ul_c2_b  = (const float*)d_in[16];
    const float* no_w     = (const float*)d_in[17];
    const float* no_b     = (const float*)d_in[18];
    float* outp = (float*)d_out;

    uintptr_t p = (uintptr_t)d_ws;
    auto take = [&](size_t bytes) -> void* {
        void* r = (void*)p;
        p += (bytes + 255) & ~(size_t)255;
        return r;
    };
    float*    xp     = (float*)take((size_t)NPOS * 4 * 4);
    float*    u      = (float*)take((size_t)NPOS * FCH * 4);
    float*    ul     = (float*)take((size_t)NPOS * FCH * 4);
    _Float16* ceU    = (_Float16*)take((size_t)NPOS * C2F * 2);
    _Float16* ceUL   = (_Float16*)take((size_t)NPOS * C2F * 2);
    _Float16* cet    = (_Float16*)take((size_t)NPOS * C2F * 2);
    float*    t1     = (float*)take((size_t)NPOS * FCH * 4);
    float*    t2     = (float*)take((size_t)NPOS * C2F * 4);
    float*    eul    = (float*)take((size_t)NPOS * FCH * 4);
    float*    params = (float*)take((size_t)NPOS * 100 * 4);

    const size_t sz_uc1  = (size_t)160 * 320 * 6;
    const size_t sz_uc2  = (size_t)320 * 320 * 6;
    const size_t sz_ulc1 = (size_t)160 * 320 * 4;
    const size_t sz_nin  = (size_t)160 * 320;
    const size_t sz_ulc2 = (size_t)320 * 320 * 4;
    _Float16 *wp_uc1[5], *wp_uc2[5], *wp_ulc1[5], *wp_nin[5], *wp_ulc2[5];
    for (int i = 0; i < 5; i++) {
        wp_uc1[i]  = (_Float16*)take(sz_uc1 * 2);
        wp_uc2[i]  = (_Float16*)take(sz_uc2 * 2);
        wp_ulc1[i] = (_Float16*)take(sz_ulc1 * 2);
        wp_nin[i]  = (_Float16*)take(sz_nin * 2);
        wp_ulc2[i] = (_Float16*)take(sz_ulc2 * 2);
    }

    const dim3 blk(256);
    const int GB160 = (NPOS * FCH + 255) / 256;

    // Input prep + tiny init convs (with shifts folded in)
    k_prep<<<(NPOS * 4 + 255) / 256, blk, 0, stream>>>(samples, xp);
    k_init_conv<<<GB160, blk, 0, stream>>>(xp, w_u_init, b_u_init, u, 2, 3, 1, 1, 1, 0, 0);
    k_init_conv<<<GB160, blk, 0, stream>>>(xp, w_ul_d, b_ul_d, ul, 1, 3, 0, 1, 1, 0, 0);
    k_init_conv<<<GB160, blk, 0, stream>>>(xp, w_ul_dr, b_ul_dr, ul, 2, 1, 1, 0, 0, 1, 1);
    k_celu<<<GB160, blk, 0, stream>>>(u, ceU);
    k_celu<<<GB160, blk, 0, stream>>>(ul, ceUL);

    // Pack all block weights to WMMA f16 fragments
    for (int i = 0; i < 5; i++) {
        k_pack<<<(int)((sz_uc1 + 255) / 256), blk, 0, stream>>>(u_c1_w + i * sz_uc1, wp_uc1[i], 160, 2, 3);
        k_pack<<<(int)((sz_uc2 + 255) / 256), blk, 0, stream>>>(u_c2_w + i * sz_uc2, wp_uc2[i], 320, 2, 3);
        k_pack<<<(int)((sz_ulc1 + 255) / 256), blk, 0, stream>>>(ul_c1_w + i * sz_ulc1, wp_ulc1[i], 160, 2, 2);
        k_pack<<<(int)((sz_nin + 255) / 256), blk, 0, stream>>>(ul_nin_w + i * sz_nin, wp_nin[i], 160, 1, 1);
        k_pack<<<(int)((sz_ulc2 + 255) / 256), blk, 0, stream>>>(ul_c2_w + i * sz_ulc2, wp_ulc2[i], 320, 2, 2);
    }

    auto conv = [&](const _Float16* cein, const _Float16* wpkp, const float* bptr,
                    float* op, int Cout, int KH, int KW, int padT, int padL, int beta) {
        int blocks = 128 * (Cout / 80);  // 256 for 160, 512 for 320
        k_conv_wmma<<<blocks, blk, 0, stream>>>(cein, wpkp, bptr, op, Cout, KH, KW,
                                                padT, padL, beta);
    };

    for (int i = 0; i < 5; i++) {
        // u-stream gated resnet (ds_conv: 2x3, padT=1 padL=1)
        conv(ceU, wp_uc1[i], u_c1_b + i * 160, t1, 160, 2, 3, 1, 1, 0);
        k_celu<<<GB160, blk, 0, stream>>>(t1, cet);
        conv(cet, wp_uc2[i], u_c2_b + i * 320, t2, 320, 2, 3, 1, 1, 0);
        k_gate<<<GB160, blk, 0, stream>>>(u, t2, ceU);   // u updated; ceU = concat_elu(u_new)
        // ul-stream gated resnet (dr_conv: 2x2, padT=1 padL=1) + NIN from updated u
        conv(ceUL, wp_ulc1[i], ul_c1_b + i * 160, t1, 160, 2, 2, 1, 1, 0);
        conv(ceU, wp_nin[i], ul_nin_b + i * 160, t1, 160, 1, 1, 0, 0, 1);  // accumulate
        k_celu<<<GB160, blk, 0, stream>>>(t1, cet);
        conv(cet, wp_ulc2[i], ul_c2_b + i * 320, t2, 320, 2, 2, 1, 1, 0);
        k_gate<<<GB160, blk, 0, stream>>>(ul, t2, ceUL);
    }

    // Output head + loss
    k_elu<<<GB160, blk, 0, stream>>>(ul, eul);
    k_ninout<<<(NPOS * 100 + 255) / 256, blk, 0, stream>>>(eul, no_w, no_b, params);
    k_zero<<<1, 32, 0, stream>>>(outp);
    k_dmll<<<NPOS / 256, blk, 0, stream>>>(samples, params, outp);
}
